// MambaDiffusionBlock_29394756174284
// MI455X (gfx1250) — matrix operand order repaired and checked by reading, not compiled, not run on hardware
//
#include <hip/hip_runtime.h>
#include <hip/hip_bf16.h>
#include <math.h>

// ---------------- problem sizes ----------------
#define BB      8
#define LL      2048
#define LTT     77
#define DD      512
#define HH      8
#define NSTATE  16
#define DCONV   4
#define DIN     1024            // d_inner = 2*D
#define DTRANK  32
#define DFF     2048
#define SIXD    (6*DD)
#define MROWS   (BB*LL)         // 16384
#define MKV     (BB*LTT)        // 616

typedef __bf16 bf16;
typedef __attribute__((ext_vector_type(16))) __bf16 v16bf;
typedef __attribute__((ext_vector_type(8)))  __bf16 v8bf;
typedef __attribute__((ext_vector_type(8)))  float  v8f;

__device__ __forceinline__ float siluf(float v)     { return v / (1.f + __expf(-v)); }
__device__ __forceinline__ float geluf(float v)     { return 0.5f * v * (1.f + erff(v * 0.70710678118654752f)); }
__device__ __forceinline__ float softplusf(float v) { return (v > 20.f) ? v : log1pf(__expf(v)); }

// ---------------- workspace layout (bytes) ----------------
static constexpr size_t al256(size_t x) { return (x + 255) & ~(size_t)255; }
static constexpr size_t OFF_WIN   = 0;                                             // [2048,512] bf16
static constexpr size_t OFF_WXDBL = al256(OFF_WIN   + (size_t)2048*512*2);         // [64,1024]
static constexpr size_t OFF_WDT   = al256(OFF_WXDBL + (size_t)64*1024*2);          // [1024,32]
static constexpr size_t OFF_WMOUT = al256(OFF_WDT   + (size_t)1024*32*2);          // [512,1024]
static constexpr size_t OFF_WQ    = al256(OFF_WMOUT + (size_t)512*1024*2);         // [512,512]
static constexpr size_t OFF_WK    = al256(OFF_WQ    + (size_t)512*512*2);
static constexpr size_t OFF_WV    = al256(OFF_WK    + (size_t)512*512*2);
static constexpr size_t OFF_WO    = al256(OFF_WV    + (size_t)512*512*2);
static constexpr size_t OFF_WF1   = al256(OFF_WO    + (size_t)512*512*2);          // [2048,512]
static constexpr size_t OFF_WF2   = al256(OFF_WF1   + (size_t)2048*512*2);         // [512,2048]
static constexpr size_t OFF_MOD   = al256(OFF_WF2   + (size_t)512*2048*2);         // [8,3072] f32
static constexpr size_t OFF_ACTBF = al256(OFF_MOD   + (size_t)BB*SIXD*4);          // [M,512] bf16 (reused 3x)
static constexpr size_t OFF_XZBF  = al256(OFF_ACTBF + (size_t)MROWS*DD*2);         // [M,2048] bf16 (also FFN h)
static constexpr size_t OFF_XCBF  = al256(OFF_XZBF  + (size_t)MROWS*2*DIN*2);      // [M,1024] bf16
static constexpr size_t OFF_DBL   = al256(OFF_XCBF  + (size_t)MROWS*DIN*2);        // [M,64] f32
static constexpr size_t OFF_DTBF  = al256(OFF_DBL   + (size_t)MROWS*64*4);         // [M,32] bf16
static constexpr size_t OFF_DELBF = al256(OFF_DTBF  + (size_t)MROWS*DTRANK*2);     // [M,1024] bf16
static constexpr size_t OFF_YGBF  = al256(OFF_DELBF + (size_t)MROWS*DIN*2);        // [M,1024] bf16
static constexpr size_t OFF_MO    = al256(OFF_YGBF  + (size_t)MROWS*DIN*2);        // [M,512] f32 (reused 3x)
static constexpr size_t OFF_QBF   = al256(OFF_MO    + (size_t)MROWS*DD*4);         // [M,512] bf16
static constexpr size_t OFF_KVBF  = al256(OFF_QBF   + (size_t)MROWS*DD*2);         // [616,512] bf16
static constexpr size_t OFF_K     = al256(OFF_KVBF  + (size_t)MKV*DD*2);           // [616,512] f32
static constexpr size_t OFF_V     = al256(OFF_K     + (size_t)MKV*DD*4);
static constexpr size_t OFF_AOBF  = al256(OFF_V     + (size_t)MKV*DD*4);           // [M,512] bf16
static constexpr size_t WS_NEED   = al256(OFF_AOBF  + (size_t)MROWS*DD*2);

// ======================================================================
// Generic bf16 WMMA GEMM: C[M,N] = act(A[M,K] @ Bt[N,K]^T + bias)
// One wave -> 32x64 strip: 2 A fragments x 4 B fragments = 8 WMMA tiles,
// so every B fragment (the dominant traffic) is reused twice.
// act: 0=none 1=gelu 2=silu 3=softplus ; outBf: 1 -> bf16 output
// ======================================================================
__global__ __launch_bounds__(256) void gemm_bf16_wmma(
    const bf16* __restrict__ A, const bf16* __restrict__ Bt,
    const float* __restrict__ bias, void* __restrict__ Cout,
    int M, int N, int K, int act, int outBf)
{
    const int lane = threadIdx.x & 31;
    const int wid  = blockIdx.x * 8 + (threadIdx.x >> 5);
    const int tilesM = (M + 31) >> 5;          // 32-row strips
    const int tilesN = N >> 6;                 // 64-col strips
    if (wid >= tilesM * tilesN) return;
    const int m0 = (wid % tilesM) << 5;
    const int n0 = (wid / tilesM) << 6;
    const int ln = lane & 15;
    const int hi = lane >> 4;                  // lane group: 0 -> low K half, 1 -> high K half

    int ar0 = m0 + ln;      if (ar0 >= M) ar0 = M - 1;   // clamp edge rows (stores predicated)
    int ar1 = m0 + 16 + ln; if (ar1 >= M) ar1 = M - 1;
    const bf16* Ap0 = A  + (size_t)ar0 * K + hi * 16;        // A frag: K {0..15} (+16 for hi)
    const bf16* Ap1 = A  + (size_t)ar1 * K + hi * 16;
    const bf16* Bp  = Bt + (size_t)(n0 + ln) * K + hi * 16;  // B frag: K {0..15} (+16 for hi)
    const size_t bstep = (size_t)16 * K;

    v8f acc[2][4] = {};
    for (int k0 = 0; k0 < K; k0 += 32) {
        v8bf a0lo = *(const v8bf*)(Ap0 + k0);
        v8bf a0hi = *(const v8bf*)(Ap0 + k0 + 8);
        v8bf a1lo = *(const v8bf*)(Ap1 + k0);
        v8bf a1hi = *(const v8bf*)(Ap1 + k0 + 8);
        v16bf a0 = __builtin_shufflevector(a0lo, a0hi, 0,1,2,3,4,5,6,7,8,9,10,11,12,13,14,15);
        v16bf a1 = __builtin_shufflevector(a1lo, a1hi, 0,1,2,3,4,5,6,7,8,9,10,11,12,13,14,15);
        v16bf b0 = *(const v16bf*)(Bp + k0);
        v16bf b1 = *(const v16bf*)(Bp + k0 + bstep);
        v16bf b2 = *(const v16bf*)(Bp + k0 + 2*bstep);
        v16bf b3 = *(const v16bf*)(Bp + k0 + 3*bstep);
        __builtin_prefetch(Ap0 + k0 + 32, 0, 1);   // -> global_prefetch_b8
        __builtin_prefetch(Ap1 + k0 + 32, 0, 1);
        __builtin_prefetch(Bp  + k0 + 32, 0, 1);
        acc[0][0] = __builtin_amdgcn_wmma_f32_16x16x32_bf16(false, a0, false, b0, (short)0, acc[0][0], false, false);
        acc[1][0] = __builtin_amdgcn_wmma_f32_16x16x32_bf16(false, a1, false, b0, (short)0, acc[1][0], false, false);
        acc[0][1] = __builtin_amdgcn_wmma_f32_16x16x32_bf16(false, a0, false, b1, (short)0, acc[0][1], false, false);
        acc[1][1] = __builtin_amdgcn_wmma_f32_16x16x32_bf16(false, a1, false, b1, (short)0, acc[1][1], false, false);
        acc[0][2] = __builtin_amdgcn_wmma_f32_16x16x32_bf16(false, a0, false, b2, (short)0, acc[0][2], false, false);
        acc[1][2] = __builtin_amdgcn_wmma_f32_16x16x32_bf16(false, a1, false, b2, (short)0, acc[1][2], false, false);
        acc[0][3] = __builtin_amdgcn_wmma_f32_16x16x32_bf16(false, a0, false, b3, (short)0, acc[0][3], false, false);
        acc[1][3] = __builtin_amdgcn_wmma_f32_16x16x32_bf16(false, a1, false, b3, (short)0, acc[1][3], false, false);
    }
#pragma unroll
    for (int t = 0; t < 2; ++t) {
#pragma unroll
        for (int j = 0; j < 4; ++j) {
            const int col = n0 + j * 16 + ln;
            const float bv = bias ? bias[col] : 0.f;
#pragma unroll
            for (int r = 0; r < 8; ++r) {
                const int row = m0 + t * 16 + r + hi * 8;  // C layout: VGPR r -> M=r (+8 hi lanes)
                if (row < M) {
                    float v = acc[t][j][r] + bv;
                    if      (act == 1) v = geluf(v);
                    else if (act == 2) v = siluf(v);
                    else if (act == 3) v = softplusf(v);
                    if (outBf) ((bf16*) Cout)[(size_t)row * N + col] = (bf16)v;
                    else       ((float*)Cout)[(size_t)row * N + col] = v;
                }
            }
        }
    }
}

// ---------------- weight transpose + f32->bf16 ----------------
__global__ __launch_bounds__(256) void transpose_w_kernel(
    const float* __restrict__ W, bf16* __restrict__ Wt, int K, int N)
{
    int idx = blockIdx.x * 256 + threadIdx.x;
    if (idx >= K * N) return;
    int k = idx / N, n = idx % N;
    Wt[(size_t)n * K + k] = (bf16)W[idx];
}

__global__ __launch_bounds__(256) void f32_to_bf16_kernel(
    const float* __restrict__ s, bf16* __restrict__ d, size_t n)
{
    size_t i = (size_t)blockIdx.x * 256 + threadIdx.x;
    if (i < n) d[i] = (bf16)s[i];
}

__global__ __launch_bounds__(256) void slice_dt_kernel(
    const float* __restrict__ dbl, bf16* __restrict__ dt)
{
    size_t i = (size_t)blockIdx.x * 256 + threadIdx.x;
    if (i >= (size_t)MROWS * DTRANK) return;
    size_t r = i >> 5; int c = (int)(i & 31);
    dt[i] = (bf16)dbl[r * 64 + c];
}

// ---------------- adaLN modulation: mod = silu(c) @ w_ada + b_ada ----------------
__global__ __launch_bounds__(256) void adaln_kernel(
    const float* __restrict__ c, const float* __restrict__ w,
    const float* __restrict__ bias, float* __restrict__ mod)
{
    int idx = blockIdx.x * 256 + threadIdx.x;
    if (idx >= BB * SIXD) return;
    int b = idx / SIXD, n = idx % SIXD;
    float acc = bias[n];
    for (int k = 0; k < DD; ++k)
        acc += siluf(c[b * DD + k]) * w[(size_t)k * SIXD + n];
    mod[idx] = acc;
}

// ---------------- LayerNorm (+ optional adaLN scale/shift), bf16 out ----------------
__global__ __launch_bounds__(256) void ln_kernel(
    const float* __restrict__ x, const float* __restrict__ mod,
    int sh_off, int sc_off, int modulate, bf16* __restrict__ out)
{
    const int lane = threadIdx.x & 31;
    const int row  = blockIdx.x * 8 + (threadIdx.x >> 5);
    if (row >= MROWS) return;
    const int b = row / LL;
    const float* xp = x + (size_t)row * DD;
    float vals[16], s = 0.f, s2 = 0.f;
#pragma unroll
    for (int i = 0; i < 16; ++i) { float v = xp[lane + 32*i]; vals[i] = v; s += v; s2 += v*v; }
#pragma unroll
    for (int off = 16; off; off >>= 1) { s += __shfl_xor(s, off); s2 += __shfl_xor(s2, off); }
    const float mean = s * (1.f / DD);
    const float var  = s2 * (1.f / DD) - mean * mean;
    const float inv  = rsqrtf(var + 1e-6f);
#pragma unroll
    for (int i = 0; i < 16; ++i) {
        const int d = lane + 32*i;
        float v = (vals[i] - mean) * inv;
        if (modulate)
            v = v * (1.f + mod[b*SIXD + sc_off + d]) + mod[b*SIXD + sh_off + d];
        out[(size_t)row * DD + d] = (bf16)v;
    }
}

// ---------------- causal depthwise conv1d + silu ----------------
__global__ __launch_bounds__(256) void conv_silu_kernel(
    const bf16* __restrict__ xz, const float* __restrict__ cw,
    const float* __restrict__ cb, bf16* __restrict__ xc)
{
    size_t i = (size_t)blockIdx.x * 256 + threadIdx.x;
    if (i >= (size_t)MROWS * DIN) return;
    const int d = (int)(i % DIN);
    const size_t row = i / DIN;
    const int l = (int)(row % LL);
    const size_t brow = row - l;       // b*LL
    float acc = cb[d];
#pragma unroll
    for (int kk = 0; kk < DCONV; ++kk) {
        const int ls = l - (DCONV - 1) + kk;
        if (ls >= 0)
            acc += cw[d * DCONV + kk] * (float)xz[(brow + ls) * (size_t)(2*DIN) + d];
    }
    xc[i] = (bf16)siluf(acc);
}

// ---------------- selective scan, fused D-skip + silu(z) gate ----------------
__global__ __launch_bounds__(256) void scan_kernel(
    const bf16* __restrict__ delta, const bf16* __restrict__ xc,
    const float* __restrict__ dbl,  const bf16* __restrict__ xz,
    const float* __restrict__ A_log, const float* __restrict__ Dskip,
    bf16* __restrict__ yg)
{
    const int ch = blockIdx.x * 256 + threadIdx.x;   // B*DIN = 8192 channels
    if (ch >= BB * DIN) return;
    const int b = ch / DIN, d = ch % DIN;
    float Arow[NSTATE], h[NSTATE];
#pragma unroll
    for (int n = 0; n < NSTATE; ++n) { Arow[n] = -__expf(A_log[d*NSTATE + n]); h[n] = 0.f; }
    const float dsk = Dskip[d];
    for (int l = 0; l < LL; ++l) {
        const size_t row = (size_t)b * LL + l;
        const float dlt = (float)delta[row * DIN + d];
        const float xv  = (float)xc[row * DIN + d];
        const float dx  = dlt * xv;
        const float* BC = dbl + row * 64;            // cols: [0,32)=dt  [32,48)=B  [48,64)=C
        float y = 0.f;
#pragma unroll
        for (int n = 0; n < NSTATE; ++n) {
            const float dA = __expf(dlt * Arow[n]);
            h[n] = dA * h[n] + dx * BC[32 + n];
            y += h[n] * BC[48 + n];
        }
        const float zv = (float)xz[row * (size_t)(2*DIN) + DIN + d];
        yg[row * DIN + d] = (bf16)((y + dsk * xv) * siluf(zv));
    }
}

// ---------------- cross-attention core: one wave per (b,h,l) query ----------------
__global__ __launch_bounds__(256) void attn_kernel(
    const bf16* __restrict__ q, const float* __restrict__ k,
    const float* __restrict__ v, bf16* __restrict__ o)
{
    const int lane = threadIdx.x & 31;
    const int w = blockIdx.x * 8 + (threadIdx.x >> 5);   // B*H*L = 131072
    if (w >= BB * HH * LL) return;
    const int b = w / (HH * LL);
    const int rem = w % (HH * LL);
    const int h = rem / LL, l = rem % LL;
    const bf16* qp = q + ((size_t)(b*LL + l)) * DD + h * 64;
    const float q0 = (float)qp[lane], q1 = (float)qp[lane + 32];
    float m = -1e30f, lsum = 0.f, a0 = 0.f, a1 = 0.f;
    for (int t = 0; t < LTT; ++t) {
        const float* kp = k + ((size_t)(b*LTT + t)) * DD + h * 64;
        float p = q0 * kp[lane] + q1 * kp[lane + 32];
#pragma unroll
        for (int off = 16; off; off >>= 1) p += __shfl_xor(p, off);
        p *= 0.125f;                                   // 1/sqrt(64)
        const float mn = fmaxf(m, p);
        const float corr = __expf(m - mn);
        const float e    = __expf(p - mn);
        const float* vp = v + ((size_t)(b*LTT + t)) * DD + h * 64;
        a0 = a0 * corr + e * vp[lane];
        a1 = a1 * corr + e * vp[lane + 32];
        lsum = lsum * corr + e;
        m = mn;
    }
    const float inv = 1.f / lsum;
    bf16* op = o + ((size_t)(b*LL + l)) * DD + h * 64;
    op[lane]      = (bf16)(a0 * inv);
    op[lane + 32] = (bf16)(a1 * inv);
}

// ---------------- residual: xout = xin + gain * mo (gain from mod or 1.0) ----------------
__global__ __launch_bounds__(256) void resid_kernel(
    const float* __restrict__ xin, const float* __restrict__ mo,
    const float* __restrict__ mod, int g_off, float* __restrict__ xout)
{
    size_t i = (size_t)blockIdx.x * 256 + threadIdx.x;
    if (i >= (size_t)MROWS * DD) return;
    const int d = (int)(i % DD);
    const int b = (int)(i / ((size_t)LL * DD));
    const float g = mod ? mod[b*SIXD + g_off + d] : 1.f;
    xout[i] = xin[i] + g * mo[i];
}

// ======================================================================
extern "C" void kernel_launch(void* const* d_in, const int* in_sizes, int n_in,
                              void* d_out, int out_size, void* d_ws, size_t ws_size,
                              hipStream_t stream)
{
    (void)in_sizes; (void)n_in; (void)out_size;
    if (ws_size < WS_NEED) return;

    const float* x      = (const float*)d_in[0];
    const float* text   = (const float*)d_in[1];
    const float* c      = (const float*)d_in[2];
    const float* w_ada  = (const float*)d_in[3];
    const float* b_ada  = (const float*)d_in[4];
    const float* w_in   = (const float*)d_in[5];
    const float* conv_w = (const float*)d_in[6];
    const float* conv_b = (const float*)d_in[7];
    const float* w_xdbl = (const float*)d_in[8];
    const float* w_dt   = (const float*)d_in[9];
    const float* b_dt   = (const float*)d_in[10];
    const float* A_log  = (const float*)d_in[11];
    const float* D_skip = (const float*)d_in[12];
    const float* w_mout = (const float*)d_in[13];
    const float* w_q = (const float*)d_in[14]; const float* b_q = (const float*)d_in[15];
    const float* w_k = (const float*)d_in[16]; const float* b_k = (const float*)d_in[17];
    const float* w_v = (const float*)d_in[18]; const float* b_v = (const float*)d_in[19];
    const float* w_o = (const float*)d_in[20]; const float* b_o = (const float*)d_in[21];
    const float* w_f1 = (const float*)d_in[22]; const float* b_f1 = (const float*)d_in[23];
    const float* w_f2 = (const float*)d_in[24]; const float* b_f2 = (const float*)d_in[25];

    char* ws = (char*)d_ws;
    bf16*  wInT   = (bf16*)(ws + OFF_WIN);
    bf16*  wXdblT = (bf16*)(ws + OFF_WXDBL);
    bf16*  wDtT   = (bf16*)(ws + OFF_WDT);
    bf16*  wMoutT = (bf16*)(ws + OFF_WMOUT);
    bf16*  wQT    = (bf16*)(ws + OFF_WQ);
    bf16*  wKT    = (bf16*)(ws + OFF_WK);
    bf16*  wVT    = (bf16*)(ws + OFF_WV);
    bf16*  wOT    = (bf16*)(ws + OFF_WO);
    bf16*  wF1T   = (bf16*)(ws + OFF_WF1);
    bf16*  wF2T   = (bf16*)(ws + OFF_WF2);
    float* modBuf = (float*)(ws + OFF_MOD);
    bf16*  actBf  = (bf16*)(ws + OFF_ACTBF);
    bf16*  xzBf   = (bf16*)(ws + OFF_XZBF);   // also reused as FFN hidden (bf16)
    bf16*  xcBf   = (bf16*)(ws + OFF_XCBF);
    float* dblBuf = (float*)(ws + OFF_DBL);
    bf16*  dtBf   = (bf16*)(ws + OFF_DTBF);
    bf16*  delBf  = (bf16*)(ws + OFF_DELBF);
    bf16*  ygBf   = (bf16*)(ws + OFF_YGBF);
    float* moBuf  = (float*)(ws + OFF_MO);    // reused: mamba-out / attn-proj / ffn-out
    bf16*  qBf    = (bf16*)(ws + OFF_QBF);
    bf16*  kvBf   = (bf16*)(ws + OFF_KVBF);
    float* kBuf   = (float*)(ws + OFF_K);
    float* vBuf   = (float*)(ws + OFF_V);
    bf16*  aoBf   = (bf16*)(ws + OFF_AOBF);

    float* xOut    = (float*)d_out;                      // residual stream, final x
    float* textOut = xOut + (size_t)MROWS * DD;

    auto eb = [](size_t n) { return (unsigned)((n + 255) / 256); };
    auto gemm = [&](const bf16* A, const bf16* Bt, const float* bias, void* C,
                    int M, int N, int K, int act, int outBf) {
        int tiles = ((M + 31) / 32) * (N / 64);
        gemm_bf16_wmma<<<(tiles + 7) / 8, 256, 0, stream>>>(A, Bt, bias, C, M, N, K, act, outBf);
    };

    // --- weight prep: f32 [K,N] -> bf16 [N,K] ---
    transpose_w_kernel<<<eb((size_t)DD*2*DIN),  256, 0, stream>>>(w_in,   wInT,   DD,  2*DIN);
    transpose_w_kernel<<<eb((size_t)DIN*64),    256, 0, stream>>>(w_xdbl, wXdblT, DIN, 64);
    transpose_w_kernel<<<eb((size_t)DTRANK*DIN),256, 0, stream>>>(w_dt,   wDtT,   DTRANK, DIN);
    transpose_w_kernel<<<eb((size_t)DIN*DD),    256, 0, stream>>>(w_mout, wMoutT, DIN, DD);
    transpose_w_kernel<<<eb((size_t)DD*DD),     256, 0, stream>>>(w_q,    wQT,    DD,  DD);
    transpose_w_kernel<<<eb((size_t)DD*DD),     256, 0, stream>>>(w_k,    wKT,    DD,  DD);
    transpose_w_kernel<<<eb((size_t)DD*DD),     256, 0, stream>>>(w_v,    wVT,    DD,  DD);
    transpose_w_kernel<<<eb((size_t)DD*DD),     256, 0, stream>>>(w_o,    wOT,    DD,  DD);
    transpose_w_kernel<<<eb((size_t)DD*DFF),    256, 0, stream>>>(w_f1,   wF1T,   DD,  DFF);
    transpose_w_kernel<<<eb((size_t)DFF*DD),    256, 0, stream>>>(w_f2,   wF2T,   DFF, DD);
    f32_to_bf16_kernel<<<eb((size_t)MKV*DD), 256, 0, stream>>>(text, kvBf, (size_t)MKV*DD);

    // --- adaLN: mod = silu(c) @ w_ada + b_ada ; offsets sh_m=0 sc_m=D g_m=2D sh_f=3D sc_f=4D g_f=5D
    adaln_kernel<<<eb((size_t)BB*SIXD), 256, 0, stream>>>(c, w_ada, b_ada, modBuf);

    // --- Mamba branch ---
    ln_kernel<<<MROWS/8, 256, 0, stream>>>(x, modBuf, /*sh*/0, /*sc*/DD, 1, actBf);
    gemm(actBf, wInT,   nullptr, xzBf,   MROWS, 2*DIN, DD,  0, 1);     // xz (bf16)
    conv_silu_kernel<<<eb((size_t)MROWS*DIN), 256, 0, stream>>>(xzBf, conv_w, conv_b, xcBf);
    gemm(xcBf,  wXdblT, nullptr, dblBuf, MROWS, 64,    DIN, 0, 0);     // [dt|B|C] (f32)
    slice_dt_kernel<<<eb((size_t)MROWS*DTRANK), 256, 0, stream>>>(dblBuf, dtBf);
    gemm(dtBf,  wDtT,   b_dt,    delBf,  MROWS, DIN, DTRANK, 3, 1);    // softplus -> delta (bf16)
    scan_kernel<<<(BB*DIN)/256, 256, 0, stream>>>(delBf, xcBf, dblBuf, xzBf, A_log, D_skip, ygBf);
    gemm(ygBf,  wMoutT, nullptr, moBuf,  MROWS, DD,  DIN, 0, 0);       // mamba out (f32)
    resid_kernel<<<eb((size_t)MROWS*DD), 256, 0, stream>>>(x, moBuf, modBuf, 2*DD, xOut);

    // --- cross-attention ---
    ln_kernel<<<MROWS/8, 256, 0, stream>>>(xOut, nullptr, 0, 0, 0, actBf);
    gemm(actBf, wQT, b_q, qBf,  MROWS, DD, DD, 0, 1);                  // q (bf16)
    gemm(kvBf,  wKT, b_k, kBuf, MKV,   DD, DD, 0, 0);                  // k (f32)
    gemm(kvBf,  wVT, b_v, vBuf, MKV,   DD, DD, 0, 0);                  // v (f32)
    attn_kernel<<<(BB*HH*LL)/8, 256, 0, stream>>>(qBf, kBuf, vBuf, aoBf);
    gemm(aoBf,  wOT, b_o, moBuf, MROWS, DD, DD, 0, 0);                 // out-proj (f32)
    resid_kernel<<<eb((size_t)MROWS*DD), 256, 0, stream>>>(xOut, moBuf, nullptr, 0, xOut);

    // --- FFN ---
    ln_kernel<<<MROWS/8, 256, 0, stream>>>(xOut, modBuf, /*sh*/3*DD, /*sc*/4*DD, 1, actBf);
    gemm(actBf, wF1T, b_f1, xzBf,  MROWS, DFF, DD, 1, 1);              // gelu -> h (bf16, reuses xz)
    gemm(xzBf,  wF2T, b_f2, moBuf, MROWS, DD, DFF, 0, 0);              // ffn out (f32)
    resid_kernel<<<eb((size_t)MROWS*DD), 256, 0, stream>>>(xOut, moBuf, modBuf, 5*DD, xOut);

    // --- second tuple output: text_emb passthrough ---
    hipMemcpyAsync(textOut, text, (size_t)MKV * DD * sizeof(float),
                   hipMemcpyDeviceToDevice, stream);
}
